// KernelPredictor_2886218023184
// MI455X (gfx1250) — matrix operand order repaired and checked by reading, not compiled, not run on hardware
//
#include <hip/hip_runtime.h>
#include <hip/hip_bf16.h>

// ---------------------------------------------------------------------------
// Fused kernel-prediction filter for gfx1250 (CDNA5, wave32, WMMA).
//   hid  = relu(data . w1^T + b1)        GEMM1: M=16 pix, K=32, N=96
//   kern = softmax(hid . w2^T + b2)      GEMM2: M=16 pix, K=96, N=441(->448)
//   out  = sum_k patch_k * kern_k        21x21 neighborhood gather, 3 chans
// One workgroup = 16-pixel horizontal strip, 128 threads (4 waves).
// ---------------------------------------------------------------------------

#define H_DIM   384
#define W_DIM   384
#define CIN     32
#define INTER   96
#define NSUP    441
#define DIAM    21
#define RADIUS  10
#define TILE_M  16
#define NT2     28              // 28 * 16 = 448 padded taps
#define PADTAPS 448
#define PATCH_W 36              // 16 + 2*10
#define PATCH_H 21
#define PATCH_SZ (PATCH_H * PATCH_W)   // 756
#define HW      (H_DIM * W_DIM)

typedef __attribute__((ext_vector_type(16))) __bf16 v16bf;
typedef __attribute__((ext_vector_type(8)))  float  v8f;

#if defined(__has_builtin)
#if __has_builtin(__builtin_amdgcn_global_load_async_to_lds_b32) && \
    __has_builtin(__builtin_amdgcn_s_wait_asynccnt)
#define USE_ASYNC_LDS 1
#endif
#endif

typedef __attribute__((address_space(1))) int gas_i32;   // global (device) int
typedef __attribute__((address_space(3))) int las_i32;   // LDS int

__global__ __launch_bounds__(128)
void kpn_fused_kernel(const float* __restrict__ p_color,
                      const float* __restrict__ p_data,
                      const float* __restrict__ p_w1,
                      const float* __restrict__ p_b1,
                      const float* __restrict__ p_w2,
                      const float* __restrict__ p_b2,
                      float* __restrict__ p_out) {
    __shared__ __bf16 s_hid[INTER * TILE_M];          // [k][m]    3 KB
    __shared__ float  s_kern[PADTAPS * TILE_M];       // [tap][m] 28 KB
    __shared__ float  s_patch[3 * PATCH_SZ];          // 9.07 KB
    __shared__ float  s_redmax[128];
    __shared__ float  s_redsum[128];
    __shared__ float  s_acc[3 * 128];

    const int tid  = threadIdx.x;
    const int wave = tid >> 5;
    const int lane = tid & 31;
    const int half = lane >> 4;      // 0 = lanes 0-15, 1 = lanes 16-31
    const int l16  = lane & 15;

    const int wg     = blockIdx.x;
    const int tile_x = wg % (W_DIM / TILE_M);
    const int row_h  = wg / (W_DIM / TILE_M);
    const int w0     = tile_x * TILE_M;

    // ---- stage zero-padded 3x21x36 color patch into LDS (used much later) --
    for (int idx = tid; idx < 3 * PATCH_SZ; idx += 128) {
        int c   = idx / PATCH_SZ;
        int rem = idx - c * PATCH_SZ;
        int r   = rem / PATCH_W;
        int col = rem - r * PATCH_W;
        int gr  = row_h + r - RADIUS;
        int gc  = w0 + col - RADIUS;
        if (gr >= 0 && gr < H_DIM && gc >= 0 && gc < W_DIM) {
#if defined(USE_ASYNC_LDS)
            __builtin_amdgcn_global_load_async_to_lds_b32(
                (gas_i32*)(p_color + c * HW + gr * W_DIM + gc),
                (las_i32*)(s_patch + idx),
                0, 0);
#else
            s_patch[idx] = p_color[c * HW + gr * W_DIM + gc];
#endif
        } else {
            s_patch[idx] = 0.f;      // zero padding
        }
    }

    // ---- GEMM1: hid = relu(data . w1^T + b1) ------------------------------
    // A fragment: A[m][k] = data[k][h][w0+m]   (16-bit A 16x32 layout)
    v16bf a1;
    {
        const float* dptr = p_data + row_h * W_DIM + w0 + l16;
        const int kbase = 8 * half;
#pragma unroll
        for (int i = 0; i < 16; ++i) {
            int k = kbase + (i < 8 ? i : i + 8);
            a1[i] = (__bf16)dptr[k * HW];
        }
    }
    // 6 N-tiles of 16 hid channels, round-robin over 4 waves
    for (int nt = wave; nt < INTER / 16; nt += 4) {
        const int n = nt * 16 + l16;
        v16bf bf1;
        {
            const float* wrow = p_w1 + n * CIN + 8 * half;    // K = 8*half + {0..7, 16..23}
#pragma unroll
            for (int i = 0; i < 16; ++i) bf1[i] = (__bf16)wrow[i < 8 ? i : i + 8];
        }
        v8f cc;
        const float bias = p_b1[n];
#pragma unroll
        for (int r = 0; r < 8; ++r) cc[r] = bias;

        v8f d = __builtin_amdgcn_wmma_f32_16x16x32_bf16(
            false, a1, false, bf1, (short)0, cc, false, false);

#pragma unroll
        for (int r = 0; r < 8; ++r) {
            float v = fmaxf(d[r], 0.f);                 // relu
            int m = r + 8 * half;                       // C/D layout: M=r+8*half
            s_hid[n * TILE_M + m] = (__bf16)v;
        }
    }
#if defined(USE_ASYNC_LDS)
    __builtin_amdgcn_s_wait_asynccnt(0);
#endif
    __syncthreads();

    // ---- GEMM2: kern = hid . w2^T + b2 (f32 accum) ------------------------
    // Hoist the three K-chunk A fragments (depend only on kc, not the tile).
    v16bf af[INTER / 32];
#pragma unroll
    for (int kc = 0; kc < INTER / 32; ++kc) {
        const int kb = kc * 32 + 8 * half;
#pragma unroll
        for (int i = 0; i < 16; ++i) {
            int k = kb + (i < 8 ? i : i + 8);
            af[kc][i] = s_hid[k * TILE_M + l16];
        }
    }

    for (int j = 0; j < NT2 / 4; ++j) {                 // 7 tiles per wave
        const int nt  = wave * (NT2 / 4) + j;
        const int tap = nt * 16 + l16;
        // Padded taps (441..447) are never read downstream; clamp to stay
        // in-bounds on w2/b2 without divergence.
        const int tapc = (tap < NSUP) ? tap : (NSUP - 1);

        v8f cc;
        const float bias = p_b2[tapc];
#pragma unroll
        for (int r = 0; r < 8; ++r) cc[r] = bias;

#pragma unroll
        for (int kc = 0; kc < INTER / 32; ++kc) {       // 3 K-chunks of 32
            v16bf bfr;
            const float* wrow = p_w2 + tapc * INTER + kc * 32 + 8 * half;
#pragma unroll
            for (int i = 0; i < 16; ++i) bfr[i] = (__bf16)wrow[i < 8 ? i : i + 8];
            cc = __builtin_amdgcn_wmma_f32_16x16x32_bf16(
                false, af[kc], false, bfr, (short)0, cc, false, false);
        }
        // [tap][pixel] layout: each lane's 8 values are contiguous -> b128 x2
#pragma unroll
        for (int r = 0; r < 8; ++r) {
            s_kern[tap * TILE_M + 8 * half + r] = cc[r];
        }
    }
    __syncthreads();

    // ---- softmax over 441 taps per pixel (8 threads / pixel) --------------
    const int p  = tid >> 3;        // pixel 0..15
    const int sl = tid & 7;         // slice 0..7 (56 taps each)

    float mx = -1e30f;
    for (int t = sl * 56; t < sl * 56 + 56; ++t)
        if (t < NSUP) mx = fmaxf(mx, s_kern[t * TILE_M + p]);
    s_redmax[tid] = mx;
    __syncthreads();

    float m8 = -1e30f;
#pragma unroll
    for (int i = 0; i < 8; ++i) m8 = fmaxf(m8, s_redmax[p * 8 + i]);

    float sum = 0.f;
    for (int t = sl * 56; t < sl * 56 + 56; ++t) {
        if (t < NSUP) {
            float e = __expf(s_kern[t * TILE_M + p] - m8);
            s_kern[t * TILE_M + p] = e;     // own (pixel, slice) cell: no race
            sum += e;
        }
    }
    s_redsum[tid] = sum;
    __syncthreads();

    // ---- apply predicted kernel to color patch ----------------------------
    float acc0 = 0.f, acc1 = 0.f, acc2 = 0.f;
    for (int t = sl; t < NSUP; t += 8) {
        float wgt = s_kern[t * TILE_M + p];
        int dx = t / DIAM;
        int dy = t - dx * DIAM;
        int pc = dx * PATCH_W + p + dy;
        acc0 += wgt * s_patch[pc];
        acc1 += wgt * s_patch[PATCH_SZ + pc];
        acc2 += wgt * s_patch[2 * PATCH_SZ + pc];
    }
    s_acc[tid]       = acc0;
    s_acc[128 + tid] = acc1;
    s_acc[256 + tid] = acc2;
    __syncthreads();

    if (tid < 48) {
        int c  = tid / 16;
        int pp = tid - c * 16;
        float tot = 0.f, den = 0.f;
#pragma unroll
        for (int i = 0; i < 8; ++i) {
            tot += s_acc[c * 128 + pp * 8 + i];
            den += s_redsum[pp * 8 + i];
        }
        p_out[c * HW + row_h * W_DIM + w0 + pp] = tot / den;
    }
}

extern "C" void kernel_launch(void* const* d_in, const int* in_sizes, int n_in,
                              void* d_out, int out_size, void* d_ws, size_t ws_size,
                              hipStream_t stream) {
    const float* color = (const float*)d_in[0];
    const float* data  = (const float*)d_in[1];
    const float* w1    = (const float*)d_in[2];
    const float* b1    = (const float*)d_in[3];
    const float* w2    = (const float*)d_in[4];
    const float* b2    = (const float*)d_in[5];
    float* out = (float*)d_out;

    dim3 grid((W_DIM / TILE_M) * H_DIM);   // 24 * 384 = 9216 blocks
    dim3 block(128);                        // 4 waves (wave32)
    kpn_fused_kernel<<<grid, block, 0, stream>>>(color, data, w1, b1, w2, b2, out);
}
